// NNConv_53644141527045
// MI455X (gfx1250) — compile-verified, matrix operand order checked by reading, not run on hardware
//
#include <hip/hip_runtime.h>

// CDNA5 / gfx1250 WMMA types
typedef __attribute__((ext_vector_type(16))) __bf16 v16bf;
typedef __attribute__((ext_vector_type(8)))  float  v8f;
typedef unsigned int uint32;

// TDM builtin argument types (6-arg form: clang-23 / therock-10.0 headers)
typedef unsigned int tdm_v4u __attribute__((ext_vector_type(4)));
typedef int          tdm_v8i __attribute__((ext_vector_type(8)));
typedef int          tdm_v4i __attribute__((ext_vector_type(4)));

// ---------------------------------------------------------------------------
// f32 -> bf16 conversions
__device__ __forceinline__ unsigned short f2bf(float f) {
  unsigned u = __float_as_uint(f);
  u += 0x7fffu + ((u >> 16) & 1u);
  return (unsigned short)(u >> 16);
}

#if __has_builtin(__builtin_amdgcn_cvt_pk_bf16_f32)
typedef __attribute__((ext_vector_type(2))) __bf16 v2bf;
__device__ __forceinline__ uint32 pack2bf(float lo, float hi) {
  v2bf p = __builtin_amdgcn_cvt_pk_bf16_f32(lo, hi);
  return __builtin_bit_cast(uint32, p);
}
#else
__device__ __forceinline__ uint32 pack2bf(float lo, float hi) {
  return (uint32)f2bf(lo) | ((uint32)f2bf(hi) << 16);
}
#endif

union FragU { uint4 q[2]; v16bf v; unsigned short s[16]; uint32 u[8]; };

// Load one 16-lane-striped bf16 WMMA operand fragment (32B per lane) from LDS
__device__ __forceinline__ v16bf ld_frag(const uint32* fragBase, int lane) {
  FragU U;
  const uint4* p = (const uint4*)(fragBase + lane * 8);
  U.q[0] = p[0];
  U.q[1] = p[1];
  return U.v;
}

// Fragment layout for 16-bit A/B operands of v_wmma_*_16x16x32_bf16:
//   lane = (g<<4) | idx16   (idx16 = row M for A / col N for B)
//   VGPR-pair j, element p:  k_local = ((j>>2)<<4) + (g<<3) + ((j&3)<<1) + p
// Inverse (even kl): g = (kl>>3)&1 ; j = ((kl>>4)&1)*4 + ((kl>>1)&3)

// ---- shared swizzle helpers (write bf16 fragment images) ------------------
__device__ __forceinline__ void fill_w2_frag(int it, const float* __restrict__ w2,
                                             uint32* dst) {     // it in [0,16384)
  int n4 = it & 255, kk = it >> 8;
  int k = kk * 2, n = n4 * 4;
  float4 r0 = *(const float4*)(w2 + k * 1024 + n);
  float4 r1 = *(const float4*)(w2 + (k + 1) * 1024 + n);
  int kc = k >> 5, kl = k & 31;
  int gg = (kl >> 3) & 1;
  int jj = ((kl >> 4) & 1) * 4 + ((kl >> 1) & 3);
  float a0[4] = {r0.x, r0.y, r0.z, r0.w};
  float a1[4] = {r1.x, r1.y, r1.z, r1.w};
#pragma unroll
  for (int q = 0; q < 4; ++q) {
    int nn = n + q, ntg = nn >> 4, fl = gg * 16 + (nn & 15);
    dst[((kc * 64 + ntg) * 32 + fl) * 8 + jj] = pack2bf(a0[q], a1[q]);
  }
}
__device__ __forceinline__ void fill_w1_frag(int it, const float* __restrict__ w1,
                                             uint32* dst) {     // it in [0,512)
  int n4 = it & 31, kk = it >> 5;
  int k = kk * 2, n = n4 * 4;
  float4 r0 = *(const float4*)(w1 + k * 128 + n);
  float4 r1 = *(const float4*)(w1 + (k + 1) * 128 + n);
  int gg = (k >> 3) & 1;
  int jj = ((k >> 4) & 1) * 4 + ((k >> 1) & 3);
  float a0[4] = {r0.x, r0.y, r0.z, r0.w};
  float a1[4] = {r1.x, r1.y, r1.z, r1.w};
#pragma unroll
  for (int q = 0; q < 4; ++q) {
    int nn = n + q, nt = nn >> 4, fl = gg * 16 + (nn & 15);
    dst[(nt * 32 + fl) * 8 + jj] = pack2bf(a0[q], a1[q]);
  }
}
__device__ __forceinline__ void fill_sq32_frag(int it, const float* __restrict__ m,
                                               uint32* dst) {   // it in [0,128): 32x32 -> 2 frags
  int n4 = it & 7, kk = it >> 3;
  int k = kk * 2, n = n4 * 4;
  float4 r0 = *(const float4*)(m + k * 32 + n);
  float4 r1 = *(const float4*)(m + (k + 1) * 32 + n);
  int gg = (k >> 3) & 1;
  int jj = ((k >> 4) & 1) * 4 + ((k >> 1) & 3);
  float a0[4] = {r0.x, r0.y, r0.z, r0.w};
  float a1[4] = {r1.x, r1.y, r1.z, r1.w};
#pragma unroll
  for (int q = 0; q < 4; ++q) {
    int nn = n + q, nt = nn >> 4, fl = gg * 16 + (nn & 15);
    dst[(nt * 32 + fl) * 8 + jj] = pack2bf(a0[q], a1[q]);
  }
}

#define EDGE_WAVES 4
#define PW_BYTES   12544                 // per-wave: xs 8192 + hstage 4096 + dst 256
#define PREP_U32   68224                 // w2s 65536 | w1s 2048 | b2f 512 | b1 128
#define PREP_BYTES (PREP_U32 * 4)        // 272896 = 34112 * 8
#define SMEM_BYTES (PREP_BYTES + EDGE_WAVES * PW_BYTES)   // 323072 <= 320KB

// ===========================================================================
// Kernel 0: pre-swizzle weights into d_ws (enables TDM bulk LDS staging)
// ===========================================================================
__global__ __launch_bounds__(256)
void nnconv_prep_kernel(const float* __restrict__ w1, const float* __restrict__ b1,
                        const float* __restrict__ w2, const float* __restrict__ b2,
                        uint32* __restrict__ ws) {
  int t0 = blockIdx.x * 256 + threadIdx.x;
  int stride = gridDim.x * 256;
  for (int it = t0; it < 16384; it += stride) fill_w2_frag(it, w2, ws);
  for (int it = t0; it < 512;  it += stride) fill_w1_frag(it, w1, ws + 65536);
  for (int it = t0; it < 128;  it += stride) fill_sq32_frag(it, b2, ws + 67584);
  for (int it = t0; it < 128;  it += stride) ws[68096 + it] = __float_as_uint(b1[it]);
}

// ===========================================================================
// Kernel 1: out[n,:] = x[n,:] @ root + bias          (writes every element)
// ===========================================================================
__global__ __launch_bounds__(128, 1)
void nnconv_root_kernel(const float* __restrict__ x,
                        const float* __restrict__ root,
                        const float* __restrict__ bias,
                        float* __restrict__ out, int N)
{
  __shared__ uint32 rs[512];
  __shared__ float  bs[32];
  const int tid = threadIdx.x, lane = tid & 31;
  const int wave = tid >> 5, g = lane >> 4, n16 = lane & 15;

  if (tid < 128) fill_sq32_frag(tid, root, rs);
  if (tid < 32) bs[tid] = bias[tid];
  __syncthreads();

  const int n0 = (blockIdx.x * 4 + wave) * 16;
  if (n0 >= N) return;
  int row = n0 + n16;
  if (row >= N) row = N - 1;

  const float* xp = x + (long long)row * 32 + g * 8;
  float4 a0 = *(const float4*)(xp);
  float4 a1 = *(const float4*)(xp + 4);
  float4 a2 = *(const float4*)(xp + 16);
  float4 a3 = *(const float4*)(xp + 20);
  FragU A;
  A.u[0] = pack2bf(a0.x, a0.y);  A.u[1] = pack2bf(a0.z, a0.w);
  A.u[2] = pack2bf(a1.x, a1.y);  A.u[3] = pack2bf(a1.z, a1.w);
  A.u[4] = pack2bf(a2.x, a2.y);  A.u[5] = pack2bf(a2.z, a2.w);
  A.u[6] = pack2bf(a3.x, a3.y);  A.u[7] = pack2bf(a3.z, a3.w);

  const bool full = (n0 + 16 <= N);
#pragma unroll
  for (int nt = 0; nt < 2; ++nt) {
    v16bf B = ld_frag(rs + nt * 256, lane);
    v8f c = {0.f, 0.f, 0.f, 0.f, 0.f, 0.f, 0.f, 0.f};
    c = __builtin_amdgcn_wmma_f32_16x16x32_bf16(false, A.v, false, B,
                                                (short)0, c, false, false);
    float bv = bs[nt * 16 + n16];
    if (full) {
#pragma unroll
      for (int r = 0; r < 8; ++r)
        out[(long long)(n0 + r + 8 * g) * 32 + nt * 16 + n16] = c[r] + bv;
    } else {
#pragma unroll
      for (int r = 0; r < 8; ++r) {
        int node = n0 + r + 8 * g;
        if (node < N)
          out[(long long)node * 32 + nt * 16 + n16] = c[r] + bv;
      }
    }
  }
}

// ===========================================================================
// Kernel 2: fused edge MLP + per-edge matmul + atomic scatter-add.
// One workgroup = 4 waves; each wave owns 64 edges per tile iteration.
// ===========================================================================
__global__ __launch_bounds__(128, 1)
void nnconv_edge_kernel(const float* __restrict__ x,
                        const long long* __restrict__ ei,
                        const float* __restrict__ ea,
                        const float* __restrict__ w1,
                        const float* __restrict__ b1,
                        const float* __restrict__ w2,
                        const float* __restrict__ b2,
                        float* __restrict__ out,
                        const uint32* __restrict__ wsrc,
                        int useTdm, int N, int E)
{
  extern __shared__ char smem[];
  uint32* w2s = (uint32*)smem;                 // 65536 u32: 256 B-frags of w2
  uint32* w1s = w2s + 65536;                   //  2048 u32:   8 B-frags of w1
  uint32* b2f = w1s + 2048;                    //   512 u32:   2 B-frags of b2 (32x32)
  float*  b1s = (float*)(b2f + 512);           //   128 f32
  char*   pw  = (char*)(b1s + 128);            // per-wave regions

  const int tid = threadIdx.x, lane = tid & 31;
  const int wave = tid >> 5, g = lane >> 4, n16 = lane & 15;

  float*          xs   = (float*)(pw + wave * PW_BYTES);                  // [32][64] x[src]^T
  unsigned short* hst  = (unsigned short*)(pw + wave * PW_BYTES + 8192);  // [16][128] bf16 h
  int*            dstl = (int*)(pw + wave * PW_BYTES + 8192 + 4096);      // [64]

  // ---------------- one-time LDS staging ---------------------------------
  bool need_fill = true;
#if __has_builtin(__builtin_amdgcn_tensor_load_to_lds) && __has_builtin(__builtin_amdgcn_s_wait_tensorcnt)
  if (useTdm) {
    if (wave == 0) {
      // 1-D TDM descriptor: 34112 x 8B elements, global(wsrc) -> LDS offset 0
      unsigned long long ga = (unsigned long long)(uintptr_t)wsrc;
      const uint32 NE8 = PREP_BYTES / 8;       // 34112
      tdm_v4u g0;
      g0[0] = 1u;                                              // count=1
      g0[1] = 0u;                                              // lds_addr
      g0[2] = (uint32)(ga & 0xffffffffu);                      // global_addr[31:0]
      g0[3] = (uint32)((ga >> 32) & 0x1ffffffu) | (2u << 30);  // addr[56:32], type=2
      tdm_v8i g1;
      g1[0] = (3 << 16);                                       // data_size = 8B
      g1[1] = (int)((NE8 & 0xffffu) << 16);                    // tensor_dim0[15:0]
      g1[2] = (int)(((NE8 >> 16) & 0xffffu) | (1u << 16));     // tensor_dim0[31:16], tensor_dim1=1
      g1[3] = (int)((NE8 & 0xffffu) << 16);                    // tile_dim0
      g1[4] = 0;                                               // tile_dim1/2 unused
      g1[5] = (int)NE8;                                        // tensor_dim0_stride lo
      g1[6] = 0;
      g1[7] = 0;
      tdm_v4i g2 = {0, 0, 0, 0}, g3 = {0, 0, 0, 0};
      tdm_v8i g4 = {0, 0, 0, 0, 0, 0, 0, 0};
      __builtin_amdgcn_tensor_load_to_lds(g0, g1, g2, g3, g4, 0);
      __builtin_amdgcn_s_wait_tensorcnt((short)0);
    }
    need_fill = false;
  }
#endif
  if (need_fill) {
    for (int it = tid; it < 16384; it += 128) fill_w2_frag(it, w2, w2s);
    for (int it = tid; it < 512;  it += 128) fill_w1_frag(it, w1, w1s);
    for (int it = tid; it < 128;  it += 128) fill_sq32_frag(it, b2, b2f);
    if (tid < 128) b1s[tid] = b1[tid];
  }
  __syncthreads();

  float b1v[8];
#pragma unroll
  for (int nt = 0; nt < 8; ++nt) b1v[nt] = b1s[nt * 16 + n16];

  const int nTiles = (E + 255) >> 8;
  for (int tile = blockIdx.x; tile < nTiles; tile += gridDim.x) {
    const long long e0 = (long long)tile * 256 + wave * 64;

    // -------- gather edge indices and x[src] rows (transposed into LDS) ---
#pragma unroll
    for (int h = 0; h < 2; ++h) {
      int m = lane + h * 32;
      long long e = e0 + m;
      if (e >= E) e = E - 1;
      int s = (int)ei[e];
      int d = (int)ei[(long long)E + e];
      dstl[m] = d;
      const float4* xp = (const float4*)(x + (long long)s * 32);
#pragma unroll
      for (int qq = 0; qq < 8; ++qq) {
        float4 v = xp[qq];
        xs[(qq * 4 + 0) * 64 + m] = v.x;
        xs[(qq * 4 + 1) * 64 + m] = v.y;
        xs[(qq * 4 + 2) * 64 + m] = v.z;
        xs[(qq * 4 + 3) * 64 + m] = v.w;
      }
    }
    if (tile + (int)gridDim.x < nTiles) {     // global_prefetch_b8 for next tile
      long long pe = (long long)(tile + gridDim.x) * 256 + wave * 64 + lane * 2;
      __builtin_prefetch(ea + pe * 32, 0, 1);
    }
    asm volatile("" ::: "memory");

    // -------- seed acc with x_src @ b2-matrix; GEMM1; stage h; A-frags ----
    v8f acc[4][2];
    v16bf B2a = ld_frag(b2f, lane);
    v16bf B2b = ld_frag(b2f + 256, lane);
    v16bf Ah[4][4];   // [sub-tile][k-chunk] of h (16 x 128, bf16)
#pragma unroll
    for (int t = 0; t < 4; ++t) {
      // x_src A-fragment straight out of the transposed LDS copy
      FragU AX;
      const int col = t * 16 + n16;
#pragma unroll
      for (int j = 0; j < 4; ++j) {
        AX.u[j]     = pack2bf(xs[(8 * g + 2 * j) * 64 + col],
                              xs[(8 * g + 2 * j + 1) * 64 + col]);
        AX.u[4 + j] = pack2bf(xs[(16 + 8 * g + 2 * j) * 64 + col],
                              xs[(16 + 8 * g + 2 * j + 1) * 64 + col]);
      }
      v8f z = {0.f, 0.f, 0.f, 0.f, 0.f, 0.f, 0.f, 0.f};
      acc[t][0] = __builtin_amdgcn_wmma_f32_16x16x32_bf16(false, AX.v, false, B2a,
                                                          (short)0, z, false, false);
      acc[t][1] = __builtin_amdgcn_wmma_f32_16x16x32_bf16(false, AX.v, false, B2b,
                                                          (short)0, z, false, false);

      long long e = e0 + t * 16 + n16;
      if (e >= E) e = E - 1;
      const float* ap = ea + e * 32 + g * 8;
      float4 a0 = *(const float4*)(ap);
      float4 a1 = *(const float4*)(ap + 4);
      float4 a2 = *(const float4*)(ap + 16);
      float4 a3 = *(const float4*)(ap + 20);
      FragU A;
      A.u[0] = pack2bf(a0.x, a0.y);  A.u[1] = pack2bf(a0.z, a0.w);
      A.u[2] = pack2bf(a1.x, a1.y);  A.u[3] = pack2bf(a1.z, a1.w);
      A.u[4] = pack2bf(a2.x, a2.y);  A.u[5] = pack2bf(a2.z, a2.w);
      A.u[6] = pack2bf(a3.x, a3.y);  A.u[7] = pack2bf(a3.z, a3.w);

#pragma unroll
      for (int nt = 0; nt < 8; ++nt) {
        v16bf Bw1 = ld_frag(w1s + nt * 256, lane);
        v8f c = {0.f, 0.f, 0.f, 0.f, 0.f, 0.f, 0.f, 0.f};
        c = __builtin_amdgcn_wmma_f32_16x16x32_bf16(false, A.v, false, Bw1,
                                                    (short)0, c, false, false);
#pragma unroll
        for (int r = 0; r < 8; ++r) {
          float hv = fmaxf(c[r] + b1v[nt], 0.0f);     // bias + ReLU
          hst[(r + 8 * g) * 128 + nt * 16 + n16] = f2bf(hv);
        }
      }
      asm volatile("" ::: "memory");   // same-wave DS ops are in-order
#pragma unroll
      for (int kc = 0; kc < 4; ++kc) {
        FragU U;
        const char* hb = (const char*)hst + n16 * 256 + kc * 64 + g * 16;
        U.q[0] = *(const uint4*)(hb);
        U.q[1] = *(const uint4*)(hb + 32);
        Ah[t][kc] = U.v;
      }
      asm volatile("" ::: "memory");
    }

    // -------- GEMM2 (h @ w2 blocks) fused with x-scaling ------------------
    for (int i = 0; i < 32; ++i) {
#pragma unroll
      for (int nt = 0; nt < 2; ++nt) {
        v16bf Bf[4];
#pragma unroll
        for (int kc = 0; kc < 4; ++kc)
          Bf[kc] = ld_frag(w2s + (kc * 64 + (i * 2 + nt)) * 256, lane);
#pragma unroll
        for (int t = 0; t < 4; ++t) {
          float4 xm0 = *(const float4*)(xs + i * 64 + t * 16 + g * 8);
          float4 xm1 = *(const float4*)(xs + i * 64 + t * 16 + g * 8 + 4);
          float xm[8] = {xm0.x, xm0.y, xm0.z, xm0.w, xm1.x, xm1.y, xm1.z, xm1.w};
          v8f cg = {0.f, 0.f, 0.f, 0.f, 0.f, 0.f, 0.f, 0.f};
#pragma unroll
          for (int kc = 0; kc < 4; ++kc)
            cg = __builtin_amdgcn_wmma_f32_16x16x32_bf16(false, Ah[t][kc], false, Bf[kc],
                                                         (short)0, cg, false, false);
#pragma unroll
          for (int r = 0; r < 8; ++r)
            acc[t][nt][r] += xm[r] * cg[r];           // pure v_fmac, co-executes
        }
      }
    }

    // -------- atomic scatter-add into out[dst] ----------------------------
#pragma unroll
    for (int t = 0; t < 4; ++t) {
#pragma unroll
      for (int r = 0; r < 8; ++r) {
        int m = t * 16 + g * 8 + r;
        long long e = e0 + m;
        int d = dstl[m];
        if (e < (long long)E) {
          float* op = out + (long long)d * 32 + n16;
          unsafeAtomicAdd(op,      acc[t][0][r]);
          unsafeAtomicAdd(op + 16, acc[t][1][r]);
        }
      }
    }
  }
}

// ===========================================================================
extern "C" void kernel_launch(void* const* d_in, const int* in_sizes, int n_in,
                              void* d_out, int out_size, void* d_ws, size_t ws_size,
                              hipStream_t stream) {
  const float*     x    = (const float*)d_in[0];
  const long long* ei   = (const long long*)d_in[1];   // int64 edge_index
  const float*     ea   = (const float*)d_in[2];
  const float*     w1   = (const float*)d_in[3];
  const float*     b1   = (const float*)d_in[4];
  const float*     w2   = (const float*)d_in[5];
  const float*     b2   = (const float*)d_in[6];
  const float*     root = (const float*)d_in[7];
  const float*     bias = (const float*)d_in[8];
  float*           out  = (float*)d_out;

  const int N = in_sizes[0] / 32;
  const int E = in_sizes[2] / 32;
  (void)n_in; (void)out_size;

  const int useTdm = (d_ws != nullptr && ws_size >= (size_t)PREP_BYTES) ? 1 : 0;

  (void)hipFuncSetAttribute(reinterpret_cast<const void*>(nnconv_edge_kernel),
                            hipFuncAttributeMaxDynamicSharedMemorySize, SMEM_BYTES);

  if (useTdm)   // pre-swizzled weight image for the TDM bulk LDS load
    nnconv_prep_kernel<<<64, 256, 0, stream>>>(w1, b1, w2, b2, (uint32*)d_ws);

  // 1) out = x @ root + bias  (initializes every output element)
  nnconv_root_kernel<<<(N + 63) / 64, 128, 0, stream>>>(x, root, bias, out, N);

  // 2) fused edge pipeline with atomic scatter-add
  int tiles = (E + 255) / 256;
  int gw = tiles < 256 ? tiles : 256;
  nnconv_edge_kernel<<<gw, 128, SMEM_BYTES, stream>>>(x, ei, ea, w1, b1, w2, b2,
                                                      out, (const uint32*)d_ws,
                                                      useTdm, N, E);
}
